// MinkUNetClassifier_31421980737579
// MI455X (gfx1250) — matrix operand order repaired and checked by reading, not compiled, not run on hardware
//
#include <hip/hip_runtime.h>
#include <hip/hip_bf16.h>

typedef __attribute__((ext_vector_type(16))) __bf16 v16bf;
typedef __attribute__((ext_vector_type(8)))  __bf16 v8bf;
typedef __attribute__((ext_vector_type(8)))  float  v8f;
typedef long long LL;

#define EPSV 1e-5f

// ---------------------------------------------------------------------------
// f32 -> bf16 convert (input, convT weights, post-conv casts)
// ---------------------------------------------------------------------------
__global__ void k_cvt_f32_bf16(const float* __restrict__ s, __bf16* __restrict__ d, LL n) {
  LL i  = (LL)blockIdx.x * blockDim.x + threadIdx.x;
  LL st = (LL)gridDim.x * blockDim.x;
  for (; i < n; i += st) d[i] = (__bf16)s[i];
}

__global__ void k_zero(float* __restrict__ p, int n) {
  int i = blockIdx.x * blockDim.x + threadIdx.x;
  if (i < n) p[i] = 0.f;
}

// ---------------------------------------------------------------------------
// Pack HWIO f32 weights into WMMA B-fragment order (bf16):
//   dst[(((cot*KB + kb)*32 + lane)*16 + e] = w[k*Cout + cot*16 + (lane&15)]
//   with k = kb*32 + (e>>3)*16 + (lane>>4)*8 + (e&7)   (zero-padded past Ktot)
// so the conv kernel loads each B fragment as one contiguous 32B vector.
// ---------------------------------------------------------------------------
__global__ void k_pack_b(const float* __restrict__ w, __bf16* __restrict__ dst,
                         int Ktot, int Cout) {
  const int KB = (Ktot + 31) / 32;
  const LL total = (LL)(Cout / 16) * KB * 32 * 16;
  for (LL i = (LL)blockIdx.x * blockDim.x + threadIdx.x; i < total;
       i += (LL)gridDim.x * blockDim.x) {
    const int e = (int)(i & 15);
    LL t = i >> 4;
    const int lane = (int)(t & 31); t >>= 5;
    const int kb  = (int)(t % KB);
    const int cot = (int)(t / KB);
    const int k = kb * 32 + (e >> 3) * 16 + (lane >> 4) * 8 + (e & 7);
    const int n = lane & 15;
    float v = 0.f;
    if (k < Ktot) v = w[(LL)k * Cout + cot * 16 + n];
    dst[i] = (__bf16)v;
  }
}

// ---------------------------------------------------------------------------
// Implicit-GEMM conv via WMMA bf16 -> f32.
// One wave computes a 16(pixel) x (NT*16)(out-ch) tile: NT accumulators share
// one A fragment per 32-K step (NT=2 whenever Cout%32==0). K-decomposition is
// compile-time; CIN%32==0 takes vectorized 16B A loads (one (kh,kw) tap per
// K-block). B is pre-packed in fragment order (contiguous 32B per lane).
// ---------------------------------------------------------------------------
template <int CIN, int KH, int KW, int NT>
__global__ __launch_bounds__(32) void k_conv_wmma(
    const __bf16* __restrict__ in, const __bf16* __restrict__ bpack,
    float* __restrict__ out,
    int N, int Hin, int Win, int Hout, int Wout, int Cout,
    int stride, int pad)
{
  constexpr int Ktot = KH * KW * CIN;
  constexpr int KB   = (Ktot + 31) / 32;

  const int lane = threadIdx.x;   // 0..31
  const int hh   = lane >> 4;
  const int ln   = lane & 15;
  const LL  P     = (LL)N * Hout * Wout;
  const LL  mbase = (LL)blockIdx.x * 16;
  const int cot0  = blockIdx.y * NT;       // first 16-wide out-ch tile

  LL pm = mbase + ln;
  int owA = 0, ohA = 0, nA = 0;
  const bool mvalid = (pm < P);
  if (mvalid) {
    owA = (int)(pm % Wout);
    LL t = pm / Wout;
    ohA = (int)(t % Hout);
    nA  = (int)(t / Hout);
  }
  const int ihb = ohA * stride - pad;
  const int iwb = owA * stride - pad;

  // per-tile B fragment base pointers
  const __bf16* bp[NT];
#pragma unroll
  for (int t = 0; t < NT; ++t)
    bp[t] = bpack + ((LL)(cot0 + t) * KB * 32 + lane) * 16;

  v8f acc[NT];
#pragma unroll
  for (int t = 0; t < NT; ++t) acc[t] = (v8f){};

#pragma unroll 2
  for (int kb = 0; kb < KB; ++kb) {
    v16bf a = {};
    if constexpr (CIN % 32 == 0) {
      // whole 32-K block sits inside one (kh,kw) tap
      const int r  = (kb * 32) / CIN;       // compile-time magic div
      const int cb = kb * 32 - r * CIN;
      const int kh = r / KW, kw = r % KW;
      const int ih = ihb + kh, iw = iwb + kw;
      if (mvalid && ih >= 0 && ih < Hin && iw >= 0 && iw < Win) {
        const __bf16* ip = in + (((LL)nA * Hin + ih) * Win + iw) * CIN + cb + hh * 8;
        v8bf lo = *(const v8bf*)(ip);
        v8bf hi = *(const v8bf*)(ip + 16);
#pragma unroll
        for (int e = 0; e < 8; ++e) { a[e] = lo[e]; a[8 + e] = hi[e]; }
      }
    } else {
#pragma unroll
      for (int e = 0; e < 16; ++e) {
        const int v  = e >> 1;
        const int kA = kb * 32 + ((v & 4) ? 16 : 0) + hh * 8 + ((v & 3) << 1) + (e & 1);
        __bf16 av = (__bf16)0.0f;
        if (mvalid && kA < Ktot) {
          const int ci = kA % CIN;           // compile-time magic div/mod
          const int r  = kA / CIN;
          const int kw = r % KW;
          const int kh = r / KW;
          const int ih = ihb + kh, iw = iwb + kw;
          if (ih >= 0 && ih < Hin && iw >= 0 && iw < Win)
            av = in[(((LL)nA * Hin + ih) * Win + iw) * CIN + ci];
        }
        a[e] = av;
      }
    }
#pragma unroll
    for (int t = 0; t < NT; ++t) {
      v16bf b = *(const v16bf*)(bp[t] + (LL)kb * 512);
      if (kb + 1 < KB) __builtin_prefetch(bp[t] + (LL)(kb + 1) * 512, 0, 1);
      acc[t] = __builtin_amdgcn_wmma_f32_16x16x32_bf16(
          false, a, false, b, (short)0, acc[t], false, false);
    }
  }
#pragma unroll
  for (int rr = 0; rr < 8; ++rr) {
    const LL pm2 = mbase + rr + hh * 8;
    if (pm2 < P) {
#pragma unroll
      for (int t = 0; t < NT; ++t)
        out[pm2 * Cout + (cot0 + t) * 16 + ln] = acc[t][rr];
    }
  }
}

// ---------------------------------------------------------------------------
// Per-channel sum / sumsq with LDS accumulation, global flush
// ---------------------------------------------------------------------------
__global__ void k_stats_f32(const float* __restrict__ x, float* __restrict__ sums,
                            LL npix, int C) {
  extern __shared__ float sm[];
  for (int i = threadIdx.x; i < 2 * C; i += blockDim.x) sm[i] = 0.f;
  __syncthreads();
  const LL total = npix * (LL)C;
  for (LL i = (LL)blockIdx.x * blockDim.x + threadIdx.x; i < total;
       i += (LL)gridDim.x * blockDim.x) {
    const int c = (int)(i % C);
    const float v = x[i];
    atomicAdd(&sm[c], v);
    atomicAdd(&sm[C + c], v * v);
  }
  __syncthreads();
  for (int i = threadIdx.x; i < 2 * C; i += blockDim.x) atomicAdd(&sums[i], sm[i]);
}

__global__ void k_stats_bf16(const __bf16* __restrict__ x, float* __restrict__ sums,
                             LL npix, int C) {
  extern __shared__ float sm2[];
  for (int i = threadIdx.x; i < 2 * C; i += blockDim.x) sm2[i] = 0.f;
  __syncthreads();
  const LL total = npix * (LL)C;
  for (LL i = (LL)blockIdx.x * blockDim.x + threadIdx.x; i < total;
       i += (LL)gridDim.x * blockDim.x) {
    const int c = (int)(i % C);
    const float v = (float)x[i];
    atomicAdd(&sm2[c], v);
    atomicAdd(&sm2[C + c], v * v);
  }
  __syncthreads();
  for (int i = threadIdx.x; i < 2 * C; i += blockDim.x) atomicAdd(&sums[i], sm2[i]);
}

// ---------------------------------------------------------------------------
// BN apply (+ optional f32 or bf16 residual) (+ optional ReLU), bf16 out
// ---------------------------------------------------------------------------
__global__ void k_bn(const float* __restrict__ x, const float* __restrict__ sums,
                     const float* __restrict__ g, const float* __restrict__ b,
                     const float* __restrict__ resF, const __bf16* __restrict__ resB,
                     int do_relu, __bf16* __restrict__ out, LL npix, int C) {
  const float inv_n = 1.0f / (float)npix;
  const LL total = npix * (LL)C;
  for (LL i = (LL)blockIdx.x * blockDim.x + threadIdx.x; i < total;
       i += (LL)gridDim.x * blockDim.x) {
    const int c = (int)(i % C);
    const float mean = sums[c] * inv_n;
    const float var  = sums[C + c] * inv_n - mean * mean;
    float y = (x[i] - mean) * rsqrtf(var + EPSV) * g[c] + b[c];
    if (resF) y += resF[i];
    if (resB) y += (float)resB[i];
    if (do_relu) y = fmaxf(y, 0.f);
    out[i] = (__bf16)y;
  }
}

// ---------------------------------------------------------------------------
// 2x2 stride-2 VALID conv-transpose into channels [0,Cout) of concat buffer
// ---------------------------------------------------------------------------
__global__ void k_convT(const __bf16* __restrict__ in, const __bf16* __restrict__ wt,
                        __bf16* __restrict__ out,
                        int N, int Hin, int Win, int Cin, int Cout, int Ctot) {
  const int Hout = Hin * 2, Wout = Win * 2;
  const LL total = (LL)N * Hout * Wout * Cout;
  for (LL i = (LL)blockIdx.x * blockDim.x + threadIdx.x; i < total;
       i += (LL)gridDim.x * blockDim.x) {
    const int co = (int)(i % Cout);
    LL p = i / Cout;
    const int ow = (int)(p % Wout); p /= Wout;
    const int oh = (int)(p % Hout);
    const int n  = (int)(p / Hout);
    const int ih = oh >> 1, di = oh & 1;
    const int iw = ow >> 1, dj = ow & 1;
    const __bf16* ip = in + (((LL)n * Hin + ih) * Win + iw) * Cin;
    const __bf16* wp = wt + ((LL)(di * 2 + dj) * Cin) * Cout + co;
    float s = 0.f;
    for (int ci = 0; ci < Cin; ++ci) s += (float)ip[ci] * (float)wp[(LL)ci * Cout];
    out[(((LL)n * Hout + oh) * Wout + ow) * Ctot + co] = (__bf16)s;
  }
}

__global__ void k_copy_skip(const __bf16* __restrict__ skip, __bf16* __restrict__ out,
                            LL npix, int Csk, int Coff, int Ctot) {
  const LL total = npix * (LL)Csk;
  for (LL i = (LL)blockIdx.x * blockDim.x + threadIdx.x; i < total;
       i += (LL)gridDim.x * blockDim.x) {
    const int c = (int)(i % Csk);
    const LL p = i / Csk;
    out[p * Ctot + Coff + c] = skip[i];
  }
}

// ---------------------------------------------------------------------------
// relu(bn(x)) then per-(batch,channel) sum (for global average pool)
// ---------------------------------------------------------------------------
__global__ void k_pool(const __bf16* __restrict__ x, const float* __restrict__ sums,
                       const float* __restrict__ g, const float* __restrict__ b,
                       float* __restrict__ pool, LL HW, int NB, int C) {
  __shared__ float sm[64];
  const int tn = NB * C;
  for (int i = threadIdx.x; i < tn; i += blockDim.x) sm[i] = 0.f;
  __syncthreads();
  const LL npix = (LL)NB * HW;
  const float inv_n = 1.0f / (float)npix;
  const LL total = npix * (LL)C;
  for (LL i = (LL)blockIdx.x * blockDim.x + threadIdx.x; i < total;
       i += (LL)gridDim.x * blockDim.x) {
    const int c = (int)(i % C);
    const LL p = i / C;
    const int n = (int)(p / HW);
    const float mean = sums[c] * inv_n;
    const float var  = sums[C + c] * inv_n - mean * mean;
    float y = ((float)x[i] - mean) * rsqrtf(var + EPSV) * g[c] + b[c];
    y = fmaxf(y, 0.f);
    atomicAdd(&sm[n * C + c], y);
  }
  __syncthreads();
  for (int i = threadIdx.x; i < tn; i += blockDim.x) atomicAdd(&pool[i], sm[i]);
}

__global__ void k_final(const float* __restrict__ pool, const float* __restrict__ lw,
                        const float* __restrict__ lb, float* __restrict__ out,
                        int NB, int C, int OC, float inv_hw) {
  const int bidx = threadIdx.x;
  if (bidx < NB) {
    for (int oc = 0; oc < OC; ++oc) {
      float s = 0.f;
      for (int c = 0; c < C; ++c) s += pool[bidx * C + c] * inv_hw * lw[c * OC + oc];
      out[bidx * OC + oc] = s + lb[oc];
    }
  }
}

// ---------------------------------------------------------------------------
// Host orchestration
// ---------------------------------------------------------------------------
extern "C" void kernel_launch(void* const* d_in, const int* in_sizes, int n_in,
                              void* d_out, int out_size, void* d_ws, size_t ws_size,
                              hipStream_t stream) {
  (void)in_sizes; (void)n_in; (void)out_size; (void)ws_size;
  auto F = [&](int i) -> const float* { return (const float*)d_in[i]; };

  const int B = 4, H0 = 512, W0 = 512;
  const LL P0 = (LL)B * H0 * W0, P1 = P0 / 4, P2 = P0 / 16, P3 = P0 / 64;

  char* base = (char*)d_ws;
  size_t off = 0;
  auto alloc = [&](size_t bytes) -> void* {
    void* p = base + off;
    off = (off + bytes + 255) & ~(size_t)255;
    return p;
  };
  auto abf  = [&](LL n) -> __bf16* { return (__bf16*)alloc((size_t)n * 2); };
  auto af32 = [&](LL n) -> float*  { return (float*)alloc((size_t)n * 4); };

  // Activation buffers
  __bf16* xb    = abf(P0 * 1);
  __bf16* h0    = abf(P0 * 16);
  __bf16* a1b   = abf(P0 * 32);   // mid-resblock activation scratch (max size)
  float*  tmpF  = af32(P0 * 32);  // conv output scratch (max size)
  float*  tmpS  = af32(P0 * 32);  // shortcut conv scratch (max size)
  __bf16* skip0 = abf(P0 * 32);
  __bf16* h1    = abf(P1 * 32);
  __bf16* skip1 = abf(P1 * 64);
  __bf16* h2    = abf(P2 * 64);
  __bf16* skip2 = abf(P2 * 128);
  __bf16* h3    = abf(P3 * 128);
  __bf16* bo    = abf(P3 * 128);
  __bf16* cat0  = abf(P2 * 192);
  __bf16* d0o   = abf(P2 * 64);
  __bf16* cat1  = abf(P1 * 96);
  __bf16* d1o   = abf(P1 * 32);
  __bf16* cat2  = abf(P0 * 48);
  __bf16* d2o   = abf(P0 * 16);
  float*  sums  = af32(512);
  float*  pool  = af32(64);

  auto cvt = [&](const float* s, __bf16* d, LL n) {
    k_cvt_f32_bf16<<<dim3(2048), 256, 0, stream>>>(s, d, n);
  };
  // Pack a conv weight (HWIO f32, input index idx) into B-fragment order.
  auto wpack = [&](int idx, int Ktot, int Cout) -> __bf16* {
    const int KB = (Ktot + 31) / 32;
    const LL n = (LL)(Cout / 16) * KB * 512;
    __bf16* d = abf(n);
    k_pack_b<<<dim3(256), 256, 0, stream>>>(F(idx), d, Ktot, Cout);
    return d;
  };

  // ---- input + weights ----------------------------------------------------
  cvt(F(0), xb, P0);
  __bf16* wc0 = wpack(1, 9 * 1, 16);  // (3,3,1,16)

  __bf16 *ew1[3], *ew2[3], *ews[3], *edn[3];
  const float *eg1[3], *eb1[3], *eg2[3], *eb2[3];
  const int echin[3] = {16, 32, 64};
  for (int l = 0; l < 3; ++l) {
    const int bi = 2 + l * 8;
    const int ci = echin[l], co = ci * 2;
    ew1[l] = wpack(bi + 0, 9 * ci, co);
    eg1[l] = F(bi + 1); eb1[l] = F(bi + 2);
    ew2[l] = wpack(bi + 3, 9 * co, co);
    eg2[l] = F(bi + 4); eb2[l] = F(bi + 5);
    ews[l] = wpack(bi + 6, ci, co);
    edn[l] = wpack(bi + 7, 4 * co, co);
  }
  __bf16* bw1 = wpack(26, 9 * 128, 128);
  const float* bg1 = F(27); const float* bb1 = F(28);
  __bf16* bw2 = wpack(29, 9 * 128, 128);
  const float* bg2 = F(30); const float* bb2 = F(31);

  __bf16 *dup[3], *dw1[3], *dw2[3], *dws[3];
  const float *dg1[3], *db1[3], *dg2[3], *db2[3];
  const int dchin[3] = {128, 64, 32};  // channels into conv_transpose
  const int dskip[3] = {128, 64, 32};  // skip channels
  for (int i2 = 0; i2 < 3; ++i2) {
    const int bi = 32 + i2 * 8;
    const int cu = dchin[i2];
    const int up = cu / 2;
    const int cc = up + dskip[i2];
    dup[i2] = abf((LL)4 * cu * up);
    cvt(F(bi + 0), dup[i2], (LL)4 * cu * up);   // convT uses plain HWIO bf16
    dw1[i2] = wpack(bi + 1, 9 * cc, up);
    dg1[i2] = F(bi + 2); db1[i2] = F(bi + 3);
    dw2[i2] = wpack(bi + 4, 9 * up, up);
    dg2[i2] = F(bi + 5); db2[i2] = F(bi + 6);
    dws[i2] = wpack(bi + 7, cc, up);
  }
  const float* g_out = F(56); const float* b_out = F(57);
  const float* lin_w = F(58); const float* lin_b = F(59);

  // ---- helpers ------------------------------------------------------------
  auto conv = [&](const __bf16* in, const __bf16* bp, float* out,
                  int N, int Hin, int Win, int Cin, int Hout, int Wout, int Cout,
                  int KH, int KW, int stride, int pad) {
    const LL P = (LL)N * Hout * Wout;
    const unsigned gx = (unsigned)((P + 15) / 16);
#define CONV_CASE(CI, KKH, KKW)                                                   \
    if (Cin == CI && KH == KKH && KW == KKW) {                                    \
      if (Cout % 32 == 0) {                                                       \
        k_conv_wmma<CI, KKH, KKW, 2><<<dim3(gx, Cout / 32), 32, 0, stream>>>(     \
            in, bp, out, N, Hin, Win, Hout, Wout, Cout, stride, pad);             \
      } else {                                                                    \
        k_conv_wmma<CI, KKH, KKW, 1><<<dim3(gx, Cout / 16), 32, 0, stream>>>(     \
            in, bp, out, N, Hin, Win, Hout, Wout, Cout, stride, pad);             \
      }                                                                           \
      return;                                                                     \
    }
    CONV_CASE(1, 3, 3)   CONV_CASE(16, 3, 3)  CONV_CASE(32, 3, 3)  CONV_CASE(48, 3, 3)
    CONV_CASE(64, 3, 3)  CONV_CASE(96, 3, 3)  CONV_CASE(128, 3, 3) CONV_CASE(192, 3, 3)
    CONV_CASE(16, 1, 1)  CONV_CASE(32, 1, 1)  CONV_CASE(48, 1, 1)  CONV_CASE(64, 1, 1)
    CONV_CASE(96, 1, 1)  CONV_CASE(192, 1, 1)
    CONV_CASE(32, 2, 2)  CONV_CASE(64, 2, 2)  CONV_CASE(128, 2, 2)
#undef CONV_CASE
  };
  auto stats = [&](const float* x, LL npix, int C) {
    k_zero<<<dim3(2), 256, 0, stream>>>(sums, 2 * C);
    k_stats_f32<<<dim3(1024), 256, (size_t)2 * C * sizeof(float), stream>>>(x, sums, npix, C);
  };
  auto statsb = [&](const __bf16* x, LL npix, int C) {
    k_zero<<<dim3(2), 256, 0, stream>>>(sums, 2 * C);
    k_stats_bf16<<<dim3(1024), 256, (size_t)2 * C * sizeof(float), stream>>>(x, sums, npix, C);
  };
  auto bn = [&](const float* x, const float* g, const float* b,
                const float* resF, const __bf16* resB, int relu,
                __bf16* out, LL npix, int C) {
    k_bn<<<dim3(2048), 256, 0, stream>>>(x, sums, g, b, resF, resB, relu, out, npix, C);
  };
  auto resblock = [&](const __bf16* xin, __bf16* yout, int Cin, int Cout,
                      int N, int H, int W,
                      const __bf16* w1b, const float* g1, const float* b1,
                      const __bf16* w2b, const float* g2, const float* b2,
                      const __bf16* wsb /* null => identity shortcut */) {
    const LL P = (LL)N * H * W;
    conv(xin, w1b, tmpF, N, H, W, Cin, H, W, Cout, 3, 3, 1, 1);
    stats(tmpF, P, Cout);
    bn(tmpF, g1, b1, nullptr, nullptr, 1, a1b, P, Cout);
    conv(a1b, w2b, tmpF, N, H, W, Cout, H, W, Cout, 3, 3, 1, 1);
    if (wsb) conv(xin, wsb, tmpS, N, H, W, Cin, H, W, Cout, 1, 1, 1, 0);
    stats(tmpF, P, Cout);
    bn(tmpF, g2, b2, wsb ? tmpS : nullptr, wsb ? nullptr : xin, 1, yout, P, Cout);
  };

  // ---- network ------------------------------------------------------------
  // stem
  conv(xb, wc0, tmpF, B, H0, W0, 1, H0, W0, 16, 3, 3, 1, 1);
  cvt(tmpF, h0, P0 * 16);

  // encoder
  resblock(h0, skip0, 16, 32, B, 512, 512, ew1[0], eg1[0], eb1[0], ew2[0], eg2[0], eb2[0], ews[0]);
  conv(skip0, edn[0], tmpF, B, 512, 512, 32, 256, 256, 32, 2, 2, 2, 0);
  cvt(tmpF, h1, P1 * 32);

  resblock(h1, skip1, 32, 64, B, 256, 256, ew1[1], eg1[1], eb1[1], ew2[1], eg2[1], eb2[1], ews[1]);
  conv(skip1, edn[1], tmpF, B, 256, 256, 64, 128, 128, 64, 2, 2, 2, 0);
  cvt(tmpF, h2, P2 * 64);

  resblock(h2, skip2, 64, 128, B, 128, 128, ew1[2], eg1[2], eb1[2], ew2[2], eg2[2], eb2[2], ews[2]);
  conv(skip2, edn[2], tmpF, B, 128, 128, 128, 64, 64, 128, 2, 2, 2, 0);
  cvt(tmpF, h3, P3 * 128);

  // bottleneck (identity shortcut)
  resblock(h3, bo, 128, 128, B, 64, 64, bw1, bg1, bb1, bw2, bg2, bb2, nullptr);

  // decoder 0: 64px -> 128px, 128ch -> 64ch, concat 128 skip => 192
  k_convT<<<dim3(2048), 256, 0, stream>>>(bo, dup[0], cat0, B, 64, 64, 128, 64, 192);
  k_copy_skip<<<dim3(2048), 256, 0, stream>>>(skip2, cat0, P2, 128, 64, 192);
  resblock(cat0, d0o, 192, 64, B, 128, 128, dw1[0], dg1[0], db1[0], dw2[0], dg2[0], db2[0], dws[0]);

  // decoder 1: 128px -> 256px, 64ch -> 32ch, concat 64 skip => 96
  k_convT<<<dim3(2048), 256, 0, stream>>>(d0o, dup[1], cat1, B, 128, 128, 64, 32, 96);
  k_copy_skip<<<dim3(2048), 256, 0, stream>>>(skip1, cat1, P1, 64, 32, 96);
  resblock(cat1, d1o, 96, 32, B, 256, 256, dw1[1], dg1[1], db1[1], dw2[1], dg2[1], db2[1], dws[1]);

  // decoder 2: 256px -> 512px, 32ch -> 16ch, concat 32 skip => 48
  k_convT<<<dim3(2048), 256, 0, stream>>>(d1o, dup[2], cat2, B, 256, 256, 32, 16, 48);
  k_copy_skip<<<dim3(2048), 256, 0, stream>>>(skip0, cat2, P0, 32, 16, 48);
  resblock(cat2, d2o, 48, 16, B, 512, 512, dw1[2], dg1[2], db1[2], dw2[2], dg2[2], db2[2], dws[2]);

  // head: relu(bn) -> global average pool -> linear
  statsb(d2o, P0, 16);
  k_zero<<<dim3(1), 64, 0, stream>>>(pool, 64);
  k_pool<<<dim3(1024), 256, 0, stream>>>(d2o, sums, g_out, b_out, pool, (LL)H0 * W0, B, 16);
  k_final<<<dim3(1), 32, 0, stream>>>(pool, lin_w, lin_b, (float*)d_out, B, 16, 1,
                                      1.0f / ((float)H0 * (float)W0));
}
